// AttentionBlock_23768349016400
// MI455X (gfx1250) — matrix operand order repaired and checked by reading, not compiled, not run on hardware
//
#include <hip/hip_runtime.h>
#include <hip/hip_bf16.h>

// ===================== feature detection =====================
#if __has_builtin(__builtin_amdgcn_global_load_async_to_lds_b128)
#define HAVE_ASYNC_LDS 1
#else
#define HAVE_ASYNC_LDS 0
#endif
#if __has_builtin(__builtin_amdgcn_s_wait_asynccnt)
#define HAVE_WAIT_ASYNC 1
#else
#define HAVE_WAIT_ASYNC 0
#endif
#if __has_builtin(__builtin_amdgcn_tensor_load_to_lds) && __has_builtin(__builtin_amdgcn_s_wait_tensorcnt)
#define HAVE_TDM 1
#else
#define HAVE_TDM 0
#endif
#if __has_include(<hip/amd_detail/amd_gfx1250_TDM.h>)
#define TDM_6ARG 1
#else
#define TDM_6ARG 0
#endif

#if HAVE_TDM && TDM_6ARG
#warning "PATH: TDM staging, 6-arg tensor_load_to_lds"
#elif HAVE_TDM
#warning "PATH: TDM staging, 5-arg tensor_load_to_lds"
#elif HAVE_ASYNC_LDS
#warning "PATH: async-to-LDS staging (no TDM builtin)"
#else
#warning "PATH: synchronous staging fallback"
#endif

// ===================== types =====================
typedef __bf16 bf16;
typedef __bf16 bf16x16 __attribute__((ext_vector_type(16)));
typedef float  f32x8   __attribute__((ext_vector_type(8)));
typedef unsigned int u32x4 __attribute__((ext_vector_type(4)));
typedef int i32x4 __attribute__((ext_vector_type(4)));
typedef int i32x8 __attribute__((ext_vector_type(8)));
typedef int v4i __attribute__((vector_size(16)));

union Frag16 { bf16x16 v; u32x4 q[2]; };

static __device__ __forceinline__ bf16 f2bf(float f) {
    unsigned int u = __float_as_uint(f);
    unsigned int r = (u + 0x7FFFu + ((u >> 16) & 1u)) >> 16;   // RNE
    unsigned short s = (unsigned short)r;
    bf16 out;
    __builtin_memcpy(&out, &s, 2);
    return out;
}

#define DIM 1024
#define SEQ 2048
#define NHEAD 16
#define HDIM 64
#define QKVN 3072
#define MLPN 4096
#define NEG_BIG (-3.0e38f)

// ===================== async copy (global -> LDS, 16 B per lane) =====================
static __device__ __forceinline__ void async_cp16(const void* g, void* l) {
#if HAVE_ASYNC_LDS
    __builtin_amdgcn_global_load_async_to_lds_b128(
        (__attribute__((address_space(1))) v4i*)(uintptr_t)(g),
        (__attribute__((address_space(3))) v4i*)(uintptr_t)(l), 0, 0);
#else
    *(u32x4*)l = *(const u32x4*)g;
#endif
}
static __device__ __forceinline__ void async_wait0() {
#if HAVE_ASYNC_LDS
#if HAVE_WAIT_ASYNC
    __builtin_amdgcn_s_wait_asynccnt(0);
#else
    asm volatile("s_wait_asynccnt 0x0" ::: "memory");
#endif
#endif
}

// ===================== TDM: 2D tile load (128 rows x 32 bf16) with LDS row padding ======
// LDS layout produced: per row 64B data + 16B pad  ==  LDSS(=40) * 2B stride.
#if HAVE_TDM
static __device__ __forceinline__ void tdm_load_tile(const bf16* gsrc, unsigned ldsOff, int K) {
    const unsigned long long ga = (unsigned long long)(uintptr_t)gsrc;
    u32x4 g0;
    g0[0] = 1u;                                   // count=1 valid descriptor
    g0[1] = ldsOff;                               // lds_addr (bytes)
    g0[2] = (unsigned)(ga & 0xFFFFFFFFu);         // global_addr[31:0]
    g0[3] = (unsigned)((ga >> 32) & 0x1FFFFFFu)   // global_addr[56:32]
          | (2u << 30);                           // type = 2 ("image")
    i32x8 g1;
    g1[0] = (1 << 16)      // data_size = 1 -> 2 bytes
          | (1 << 20)      // pad_enable
          | (3 << 22)      // pad_interval: 3 -> every 16 DWORDs (64 B)
          | (3 << 25);     // pad_amount:   3 -> 4 DWORDs (16 B)
    g1[1] = (K & 0xFFFF) << 16;                   // tensor_dim0[15:0]  (bits 63:48)
    g1[2] = ((K >> 16) & 0xFFFF);                 // tensor_dim0[31:16]; tensor_dim1 lo16 = 0
    g1[3] = 0x10 | (32 << 16);                    // tensor_dim1 = 1<<20 rows; tile_dim0 = 32
    g1[4] = 128;                                  // tile_dim1 = 128; tile_dim2 = 0
    g1[5] = K;                                    // tensor_dim0_stride[31:0] (elements)
    g1[6] = 0;                                    // stride hi, tensor_dim1_stride lo
    g1[7] = 0;
    const i32x4 z4 = {0, 0, 0, 0};
#if TDM_6ARG
    const i32x8 z8 = {0, 0, 0, 0, 0, 0, 0, 0};
    __builtin_amdgcn_tensor_load_to_lds(g0, g1, z4, z4, z8, 0);
#else
    __builtin_amdgcn_tensor_load_to_lds(g0, g1, z4, z4, 0);
#endif
}
static __device__ __forceinline__ void tensor_wait0() {
    __builtin_amdgcn_s_wait_tensorcnt(0);
}
#endif

// ===================== fused fp32->bf16 convert + transpose: W[K][N] -> Wt[N][K] ==========
__global__ __launch_bounds__(256) void cvt_transpose_kernel(const float* __restrict__ src,
                                                            bf16* __restrict__ dst,
                                                            int K, int N) {
    __shared__ float tile[32][33];
    const int kb = blockIdx.y * 32;
    const int nb = blockIdx.x * 32;
    const int tid = threadIdx.x;
    {
        const int tk  = tid >> 3;
        const int tn4 = (tid & 7) * 4;
        const float4 v = *(const float4*)(src + (size_t)(kb + tk) * N + nb + tn4);
        tile[tk][tn4 + 0] = v.x;
        tile[tk][tn4 + 1] = v.y;
        tile[tk][tn4 + 2] = v.z;
        tile[tk][tn4 + 3] = v.w;
    }
    __syncthreads();
    {
        const int tn  = tid >> 3;
        const int tk4 = (tid & 7) * 4;
        bf16* d = dst + (size_t)(nb + tn) * K + kb + tk4;
        d[0] = f2bf(tile[tk4 + 0][tn]);
        d[1] = f2bf(tile[tk4 + 1][tn]);
        d[2] = f2bf(tile[tk4 + 2][tn]);
        d[3] = f2bf(tile[tk4 + 3][tn]);
    }
}

// ===================== LayerNorm (fp32 in -> bf16 out), one block per 1024-row =====================
__global__ __launch_bounds__(256) void layernorm_bf16_kernel(const float* __restrict__ x,
                                                             const float* __restrict__ w,
                                                             const float* __restrict__ b,
                                                             bf16* __restrict__ out) {
    __shared__ float red[16];
    const int row  = blockIdx.x;
    const int tid  = threadIdx.x;
    const int lane = tid & 31;
    const int wave = tid >> 5;
    const float4 v = ((const float4*)(x + (size_t)row * DIM))[tid];
    float s  = v.x + v.y + v.z + v.w;
    float s2 = v.x*v.x + v.y*v.y + v.z*v.z + v.w*v.w;
    #pragma unroll
    for (int off = 16; off >= 1; off >>= 1) {
        s  += __shfl_xor(s,  off, 32);
        s2 += __shfl_xor(s2, off, 32);
    }
    if (lane == 0) { red[wave] = s; red[8 + wave] = s2; }
    __syncthreads();
    float S = 0.f, S2 = 0.f;
    #pragma unroll
    for (int i = 0; i < 8; ++i) { S += red[i]; S2 += red[8 + i]; }
    const float mu   = S * (1.0f / DIM);
    const float var  = S2 * (1.0f / DIM) - mu * mu;
    const float rstd = rsqrtf(var + 1e-5f);
    const int col = tid * 4;
    bf16* o = out + (size_t)row * DIM + col;
    o[0] = f2bf((v.x - mu) * rstd * w[col + 0] + b[col + 0]);
    o[1] = f2bf((v.y - mu) * rstd * w[col + 1] + b[col + 1]);
    o[2] = f2bf((v.z - mu) * rstd * w[col + 2] + b[col + 2]);
    o[3] = f2bf((v.w - mu) * rstd * w[col + 3] + b[col + 3]);
}

// ===================== bf16 WMMA GEMM (Wt pre-transposed [N][K]), TDM/async staged =========
// EPI: 0 = bias -> bf16 out ; 1 = bias + exact GELU -> bf16 out ; 2 = bias + resid -> f32 out
#define BM 128
#define BN 128
#define BK 32
#define LDSS 40   // padded stride (elems); 80 B row keeps 16-byte alignment for b128 frag loads

template <int EPI>
__global__ __launch_bounds__(256) void gemm_bf16_kernel(const bf16* __restrict__ A,
                                                        const bf16* __restrict__ Wt,
                                                        const float* __restrict__ bias,
                                                        const float* __restrict__ resid,
                                                        void* __restrict__ out,
                                                        int M, int N, int K) {
    __shared__ bf16 sA[2][BM * LDSS];
    __shared__ bf16 sB[2][BN * LDSS];   // sB[buf][n*LDSS + k]  (K-contiguous per n)

    const int tid  = threadIdx.x;
    const int lane = tid & 31;
    const int wave = tid >> 5;     // 0..7
    const int wm   = wave >> 1;    // 0..3
    const int wn   = wave & 1;     // 0..1
    const int ln   = lane & 15;
    const int lg   = lane >> 4;

    const int rowBase = blockIdx.y * BM;
    const int colBase = blockIdx.x * BN;

#if HAVE_TDM
    // one wave drives the Tensor Data Mover: 2 descriptors (A tile, B tile) per K-step
    auto stage = [&](int k0, int buf) {
        if (wave == 0) {
            tdm_load_tile(A  + (size_t)rowBase * K + k0, (unsigned)(uintptr_t)&sA[buf][0], K);
            tdm_load_tile(Wt + (size_t)colBase * K + k0, (unsigned)(uintptr_t)&sB[buf][0], K);
        }
    };
    auto stage_wait = [&]() { if (wave == 0) tensor_wait0(); };
#else
    const int srow = tid >> 1;
    const int sh   = (tid & 1) * 16;
    auto stage = [&](int k0, int buf) {
        const bf16* sa = A  + (size_t)(rowBase + srow) * K + k0 + sh;
        const bf16* sb = Wt + (size_t)(colBase + srow) * K + k0 + sh;
        bf16* da = &sA[buf][srow * LDSS + sh];
        bf16* db = &sB[buf][srow * LDSS + sh];
        async_cp16(sa,     da);
        async_cp16(sa + 8, da + 8);
        async_cp16(sb,     db);
        async_cp16(sb + 8, db + 8);
    };
    auto stage_wait = [&]() { async_wait0(); };
#endif

    f32x8 acc[2][4];
    #pragma unroll
    for (int mt = 0; mt < 2; ++mt)
        #pragma unroll
        for (int nt = 0; nt < 4; ++nt)
            #pragma unroll
            for (int r = 0; r < 8; ++r) acc[mt][nt][r] = 0.0f;

    stage(0, 0);
    stage_wait();
    __syncthreads();

    int buf = 0;
    for (int k0 = 0; k0 < K; k0 += BK) {
        const int nxt = k0 + BK;
        if (nxt < K) stage(nxt, buf ^ 1);   // uniform condition: prefetch next tile

        Frag16 a[2], b[4];
        #pragma unroll
        for (int mt = 0; mt < 2; ++mt) {
            const int row = wm * 32 + mt * 16 + ln;
            a[mt].q[0] = *(const u32x4*)(&sA[buf][row * LDSS + lg * 8]);
            a[mt].q[1] = *(const u32x4*)(&sA[buf][row * LDSS + 16 + lg * 8]);
        }
        #pragma unroll
        for (int nt = 0; nt < 4; ++nt) {
            const int n = wn * 64 + nt * 16 + ln;
            b[nt].q[0] = *(const u32x4*)(&sB[buf][n * LDSS + lg * 8]);
            b[nt].q[1] = *(const u32x4*)(&sB[buf][n * LDSS + 16 + lg * 8]);
        }
        #pragma unroll
        for (int mt = 0; mt < 2; ++mt)
            #pragma unroll
            for (int nt = 0; nt < 4; ++nt)
                acc[mt][nt] = __builtin_amdgcn_wmma_f32_16x16x32_bf16(
                    false, a[mt].v, false, b[nt].v, (short)0, acc[mt][nt], false, false);

        stage_wait();      // prefetch into buf^1 has landed (TENSORcnt / ASYNCcnt)
        __syncthreads();   // everyone done reading buf / writing buf^1
        buf ^= 1;
    }

    // epilogue: element (m,n) of a 16x16 C tile lives in VGPR r, m = r + 8*lg, n = ln
    #pragma unroll
    for (int mt = 0; mt < 2; ++mt) {
        #pragma unroll
        for (int nt = 0; nt < 4; ++nt) {
            const int col = colBase + wn * 64 + nt * 16 + ln;
            const float bs = bias[col];
            #pragma unroll
            for (int r = 0; r < 8; ++r) {
                const int row = rowBase + wm * 32 + mt * 16 + r + 8 * lg;
                float v = acc[mt][nt][r] + bs;
                const size_t idx = (size_t)row * N + col;
                if (EPI == 2) {
                    ((float*)out)[idx] = v + resid[idx];
                } else if (EPI == 1) {
                    v = 0.5f * v * (1.0f + erff(v * 0.70710678118654752f));
                    ((bf16*)out)[idx] = f2bf(v);
                } else {
                    ((bf16*)out)[idx] = f2bf(v);
                }
            }
        }
    }
}

// ===================== Flash attention: per (b, h, 64-row q tile), 4 waves =====================
__global__ __launch_bounds__(128) void flash_attn_kernel(const bf16* __restrict__ qkv,
                                                         bf16* __restrict__ attn_out) {
    __shared__ bf16 sV[HDIM * 32];      // Vt[d][k], stride 32
    __shared__ bf16 sP[4][16 * 32];     // per-wave P tile, stride 32

    const int tid  = threadIdx.x;
    const int lane = tid & 31;
    const int wave = tid >> 5;          // 0..3
    const int ln   = lane & 15;
    const int lg   = lane >> 4;
    const int qt = blockIdx.x;          // 0..31
    const int h  = blockIdx.y;          // 0..15
    const int bb = blockIdx.z;          // 0..3

    const size_t tokBase = (size_t)bb * SEQ;
    const int qbase = qt * 64 + wave * 16;

    // Q fragments in A layout, straight from global (d contiguous)
    Frag16 qf[2];
    {
        const bf16* qp = qkv + (tokBase + qbase + ln) * QKVN + h * HDIM;
        #pragma unroll
        for (int ks = 0; ks < 2; ++ks) {
            qf[ks].q[0] = *(const u32x4*)(qp + ks * 32 + lg * 8);
            qf[ks].q[1] = *(const u32x4*)(qp + ks * 32 + 16 + lg * 8);
        }
    }

    f32x8 o[4];
    #pragma unroll
    for (int nt = 0; nt < 4; ++nt)
        #pragma unroll
        for (int r = 0; r < 8; ++r) o[nt][r] = 0.0f;
    float mrow[8], lrow[8];
    #pragma unroll
    for (int r = 0; r < 8; ++r) { mrow[r] = NEG_BIG; lrow[r] = 0.0f; }

    const int nkt = qt * 2 + 2;   // causal: uniform per block
    for (int kt = 0; kt < nkt; ++kt) {
        const int kcb = kt * 32;
        { // stage V tile transposed: sV[d*32 + krow]
            const int kr = tid >> 2;
            const int d0 = (tid & 3) * 16;
            const bf16* vp = qkv + (tokBase + kcb + kr) * QKVN + 2 * DIM + h * HDIM + d0;
            const u32x4 w0 = *(const u32x4*)(vp);
            const u32x4 w1 = *(const u32x4*)(vp + 8);
            bf16 tmp[16];
            *(u32x4*)(tmp)     = w0;
            *(u32x4*)(tmp + 8) = w1;
            #pragma unroll
            for (int i = 0; i < 16; ++i) sV[(d0 + i) * 32 + kr] = tmp[i];
        }
        __syncthreads();

        // S = Q * K^T  (K^T operand straight from global: contract dim d is contiguous)
        f32x8 s[2];
        #pragma unroll
        for (int nI = 0; nI < 2; ++nI)
            #pragma unroll
            for (int r = 0; r < 8; ++r) s[nI][r] = 0.0f;
        #pragma unroll
        for (int nI = 0; nI < 2; ++nI) {
            const int krow = kcb + nI * 16 + ln;
            const bf16* kp = qkv + (tokBase + krow) * QKVN + DIM + h * HDIM;
            #pragma unroll
            for (int ks = 0; ks < 2; ++ks) {
                Frag16 kf;
                kf.q[0] = *(const u32x4*)(kp + ks * 32 + lg * 8);
                kf.q[1] = *(const u32x4*)(kp + ks * 32 + 16 + lg * 8);
                s[nI] = __builtin_amdgcn_wmma_f32_16x16x32_bf16(
                    false, qf[ks].v, false, kf.v, (short)0, s[nI], false, false);
            }
        }

        // scale + causal mask
        #pragma unroll
        for (int nI = 0; nI < 2; ++nI) {
            const int kc = kcb + nI * 16 + ln;
            #pragma unroll
            for (int r = 0; r < 8; ++r) {
                const int qr = qbase + r + 8 * lg;
                const float v = s[nI][r] * 0.125f;
                s[nI][r] = (kc <= qr) ? v : NEG_BIG;
            }
        }

        // online softmax (row reductions across the 16-lane half)
        float alpha[8];
        #pragma unroll
        for (int r = 0; r < 8; ++r) {
            float t = fmaxf(s[0][r], s[1][r]);
            #pragma unroll
            for (int off = 8; off >= 1; off >>= 1) t = fmaxf(t, __shfl_xor(t, off, 32));
            const float mn = fmaxf(mrow[r], t);
            const float a  = __expf(mrow[r] - mn);
            mrow[r] = mn;
            const float p0 = __expf(s[0][r] - mn);
            const float p1 = __expf(s[1][r] - mn);
            s[0][r] = p0; s[1][r] = p1;
            float rs = p0 + p1;
            #pragma unroll
            for (int off = 8; off >= 1; off >>= 1) rs += __shfl_xor(rs, off, 32);
            lrow[r] = lrow[r] * a + rs;
            alpha[r] = a;
        }
        #pragma unroll
        for (int nt = 0; nt < 4; ++nt)
            #pragma unroll
            for (int r = 0; r < 8; ++r) o[nt][r] *= alpha[r];

        // C-layout -> A-layout for P via LDS
        bf16* pp = sP[wave];
        #pragma unroll
        for (int nI = 0; nI < 2; ++nI)
            #pragma unroll
            for (int r = 0; r < 8; ++r)
                pp[(r + 8 * lg) * 32 + nI * 16 + ln] = f2bf(s[nI][r]);
        __syncthreads();

        Frag16 pf;
        pf.q[0] = *(const u32x4*)(pp + ln * 32 + lg * 8);
        pf.q[1] = *(const u32x4*)(pp + ln * 32 + 16 + lg * 8);
        #pragma unroll
        for (int nt = 0; nt < 4; ++nt) {
            Frag16 vf;
            const int d = nt * 16 + ln;
            vf.q[0] = *(const u32x4*)(&sV[d * 32 + lg * 8]);
            vf.q[1] = *(const u32x4*)(&sV[d * 32 + 16 + lg * 8]);
            o[nt] = __builtin_amdgcn_wmma_f32_16x16x32_bf16(
                false, pf.v, false, vf.v, (short)0, o[nt], false, false);
        }
        __syncthreads();
    }

    // finalize + store
    #pragma unroll
    for (int r = 0; r < 8; ++r) lrow[r] = 1.0f / lrow[r];
    #pragma unroll
    for (int nt = 0; nt < 4; ++nt)
        #pragma unroll
        for (int r = 0; r < 8; ++r) {
            const int m = r + 8 * lg;
            attn_out[(tokBase + qbase + m) * DIM + h * HDIM + nt * 16 + ln] =
                f2bf(o[nt][r] * lrow[r]);
        }
}

// ===================== host =====================
extern "C" void kernel_launch(void* const* d_in, const int* in_sizes, int n_in,
                              void* d_out, int out_size, void* d_ws, size_t ws_size,
                              hipStream_t stream) {
    (void)in_sizes; (void)n_in; (void)out_size; (void)ws_size;
    const float* x      = (const float*)d_in[0];
    const float* ln1_w  = (const float*)d_in[1];
    const float* ln1_b  = (const float*)d_in[2];
    const float* qkv_w  = (const float*)d_in[3];
    const float* qkv_b  = (const float*)d_in[4];
    const float* proj_w = (const float*)d_in[5];
    const float* proj_b = (const float*)d_in[6];
    const float* ln2_w  = (const float*)d_in[7];
    const float* ln2_b  = (const float*)d_in[8];
    const float* fc1_w  = (const float*)d_in[9];
    const float* fc1_b  = (const float*)d_in[10];
    const float* fc2_w  = (const float*)d_in[11];
    const float* fc2_b  = (const float*)d_in[12];
    float* out = (float*)d_out;

    const int M = 4 * SEQ;  // 8192 tokens
    char* ws = (char*)d_ws;
    size_t off = 0;
    auto carve = [&](size_t bytes) { char* p = ws + off; off = (off + bytes + 255) & ~(size_t)255; return p; };
    bf16* wqkv = (bf16*)carve((size_t)DIM * QKVN * 2);   // transposed [3072][1024]
    bf16* wprj = (bf16*)carve((size_t)DIM * DIM * 2);    // transposed [1024][1024]
    bf16* wfc1 = (bf16*)carve((size_t)DIM * MLPN * 2);   // transposed [4096][1024]
    bf16* wfc2 = (bf16*)carve((size_t)MLPN * DIM * 2);   // transposed [1024][4096]
    bf16* lnx  = (bf16*)carve((size_t)M * DIM * 2);
    bf16* qkvb = (bf16*)carve((size_t)M * QKVN * 2);
    bf16* attn = (bf16*)carve((size_t)M * DIM * 2);
    bf16* ln2o = (bf16*)carve((size_t)M * DIM * 2);
    bf16* hbuf = (bf16*)carve((size_t)M * MLPN * 2);

    auto cvtT = [&](const float* s, bf16* d, int K, int N) {
        cvt_transpose_kernel<<<dim3(N / 32, K / 32), 256, 0, stream>>>(s, d, K, N);
    };
    cvtT(qkv_w, wqkv, DIM, QKVN);
    cvtT(proj_w, wprj, DIM, DIM);
    cvtT(fc1_w, wfc1, DIM, MLPN);
    cvtT(fc2_w, wfc2, MLPN, DIM);

    // 1) ln1(x) -> bf16
    layernorm_bf16_kernel<<<M, 256, 0, stream>>>(x, ln1_w, ln1_b, lnx);
    // 2) qkv = ln1 @ qkv_w + qkv_b  (bf16 out)
    gemm_bf16_kernel<0><<<dim3(QKVN / BN, M / BM), 256, 0, stream>>>(
        lnx, wqkv, qkv_b, nullptr, qkvb, M, QKVN, DIM);
    // 3) causal flash attention
    flash_attn_kernel<<<dim3(SEQ / 64, NHEAD, 4), 128, 0, stream>>>(qkvb, attn);
    // 4) x1 = x + attn @ proj_w + proj_b  (fp32 -> d_out)
    gemm_bf16_kernel<2><<<dim3(DIM / BN, M / BM), 256, 0, stream>>>(
        attn, wprj, proj_b, x, out, M, DIM, DIM);
    // 5) ln2(x1) -> bf16
    layernorm_bf16_kernel<<<M, 256, 0, stream>>>(out, ln2_w, ln2_b, ln2o);
    // 6) h = gelu(ln2 @ fc1_w + fc1_b)  (bf16 out)
    gemm_bf16_kernel<1><<<dim3(MLPN / BN, M / BM), 256, 0, stream>>>(
        ln2o, wfc1, fc1_b, nullptr, hbuf, M, MLPN, DIM);
    // 7) out = x1 + h @ fc2_w + fc2_b   (fp32, residual = d_out in place)
    gemm_bf16_kernel<2><<<dim3(DIM / BN, M / BM), 256, 0, stream>>>(
        hbuf, wfc2, fc2_b, out, out, M, DIM, MLPN);
}
